// QuestionAnswering_67559835566516
// MI455X (gfx1250) — matrix operand order repaired and checked by reading, not compiled, not run on hardware
//
#include <hip/hip_runtime.h>
#include <cstdint>
#include <cstddef>

// ---------------- WMMA types (CDNA5 wave32) ----------------
typedef __attribute__((ext_vector_type(16))) _Float16 v16h;
typedef __attribute__((ext_vector_type(8)))  _Float16 v8h;
typedef __attribute__((ext_vector_type(8)))  float    v8f;

// problem constants
constexpr int BATCH = 4;
constexpr int SEQL  = 2048;
constexpr int HDIM  = 768;
constexpr int NTBB  = 20;              // BATCH * TB_SIZE
constexpr int TLEN  = 512;
constexpr int MROWS = NTBB * TLEN;     // 10240
constexpr int H3    = 2304;
constexpr int FFDIM = 3072;
constexpr int NHEAD = 12;
constexpr int MSEG  = 512;
constexpr int OUTR  = BATCH * SEQL;    // 8192

#if __has_builtin(__builtin_amdgcn_global_load_async_to_lds_b128)
#define USE_ASYNC_LDS 1
typedef int v4i_vs __attribute__((__vector_size__(4 * sizeof(int))));
typedef __attribute__((address_space(1))) v4i_vs* g_v4p;   // global (AS1) int4*
typedef __attribute__((address_space(3))) v4i_vs* l_v4p;   // LDS    (AS3) int4*
#else
#define USE_ASYNC_LDS 0
#endif

__device__ __forceinline__ float gelu_f(float x) {
  return 0.5f * x * (1.0f + erff(x * 0.70710678118654752f));
}

__device__ __forceinline__ v16h mk16(v8h lo, v8h hi) {
  v16h r;
#pragma unroll
  for (int i = 0; i < 8; ++i) { r[i] = lo[i]; r[i + 8] = hi[i]; }
  return r;
}

// 16-byte global -> LDS copy; async DMA path on toolchains that expose it.
__device__ __forceinline__ void copy16_g2l(const _Float16* __restrict__ g,
                                           _Float16* __restrict__ l) {
#if USE_ASYNC_LDS
  __builtin_amdgcn_global_load_async_to_lds_b128((g_v4p)g, (l_v4p)l, 0, 0);
#else
  *(v8h*)l = *(const v8h*)g;
#endif
}

__device__ __forceinline__ void async_fence() {
#if USE_ASYNC_LDS
#if __has_builtin(__builtin_amdgcn_s_wait_asynccnt)
  __builtin_amdgcn_s_wait_asynccnt(0);
#else
  asm volatile("s_wait_asynccnt 0x0" ::: "memory");
#endif
#endif
}

// =====================================================================
// Tiled WMMA GEMM:  C[M,N] = act(A[M,K] @ W[K,N] + bias)
// A fp32 (row stride lda); Wt = W transposed, f16 [N][K]; C fp32.
// Block = 256 thr (8 waves), tile 128x128, K-step 64 (16 WMMA / stage).
// M = gridDim.x*128, N = gridDim.y*128, K % 64 == 0 (all dims satisfy).
// =====================================================================
__global__ __launch_bounds__(256)
void k_gemm(const float* __restrict__ A, int lda,
            const _Float16* __restrict__ Wt,
            const float* __restrict__ bias,
            float* __restrict__ C, int ldc,
            int N, int K, int act)
{
  __shared__ __align__(16) _Float16 As[128 * 64];   // [m][k]  16 KB
  __shared__ __align__(16) _Float16 Bs[128 * 64];   // [n][k]  16 KB
  const int tid  = threadIdx.x;
  const int lane = tid & 31, wid = tid >> 5;
  const int wm = wid & 3, wn = wid >> 2;            // 4 waves along M, 2 along N
  const int m0 = blockIdx.x * 128, n0 = blockIdx.y * 128;
  const int koff_a = (lane >= 16) ? 8 : 0;
  const int koff_b = (lane >= 16) ? 16 : 0;
  const int nrow   = lane & 15;
  const int roff   = (lane >= 16) ? 8 : 0;

  v8f acc[2][4] = {};

  for (int k0 = 0; k0 < K; k0 += 64) {
    // prefetch next tiles into cache while we stage/compute this one
    if (k0 + 64 < K) {
      __builtin_prefetch(A + (size_t)(m0 + (tid & 127)) * lda + k0 + 64, 0, 0);
      __builtin_prefetch(Wt + (size_t)(n0 + (tid & 127)) * K + k0 + 64, 0, 0);
    }
    // stage A tile (fp32 -> f16): 128 x 64
#pragma unroll
    for (int it = 0; it < 8; ++it) {
      int idx = tid + it * 256;           // 2048 float4 groups
      int r = idx >> 4, c4 = (idx & 15) * 4;
      const float4 f = *(const float4*)(A + (size_t)(m0 + r) * lda + k0 + c4);
      _Float16* d = As + r * 64 + c4;
      d[0] = (_Float16)f.x; d[1] = (_Float16)f.y;
      d[2] = (_Float16)f.z; d[3] = (_Float16)f.w;
    }
    // stage B tile (f16, pre-transposed, contiguous 16B chunks): 128 x 64
#pragma unroll
    for (int it = 0; it < 4; ++it) {
      int idx = tid + it * 256;           // 1024 16-byte chunks
      int r = idx >> 3, c8 = (idx & 7) * 8;
      copy16_g2l(Wt + (size_t)(n0 + r) * K + k0 + c8, Bs + r * 64 + c8);
    }
    async_fence();
    __syncthreads();

#pragma unroll
    for (int ks = 0; ks < 2; ++ks) {
      v16h af[2], bf[4];
#pragma unroll
      for (int tm = 0; tm < 2; ++tm) {
        const _Float16* p = As + (wm * 32 + tm * 16 + nrow) * 64 + ks * 32;
        af[tm] = mk16(*(const v8h*)(p + koff_a), *(const v8h*)(p + 16 + koff_a));
      }
#pragma unroll
      for (int tn = 0; tn < 4; ++tn) {
        const _Float16* p = Bs + (wn * 64 + tn * 16 + nrow) * 64 + ks * 32;
        bf[tn] = mk16(*(const v8h*)(p + koff_a), *(const v8h*)(p + 16 + koff_a));
      }
#pragma unroll
      for (int tm = 0; tm < 2; ++tm)
#pragma unroll
        for (int tn = 0; tn < 4; ++tn)
          acc[tm][tn] = __builtin_amdgcn_wmma_f32_16x16x32_f16(
              false, af[tm], false, bf[tn], (short)0, acc[tm][tn], false, false);
    }
    __syncthreads();
  }

#pragma unroll
  for (int tm = 0; tm < 2; ++tm) {
#pragma unroll
    for (int tn = 0; tn < 4; ++tn) {
      int gn = n0 + wn * 64 + tn * 16 + nrow;
      float bb = bias ? bias[gn] : 0.0f;
      int gmb = m0 + wm * 32 + tm * 16 + roff;
#pragma unroll
      for (int r = 0; r < 8; ++r) {
        float v = acc[tm][tn][r] + bb;
        if (act) v = gelu_f(v);
        C[(size_t)(gmb + r) * ldc + gn] = v;
      }
    }
  }
}

// =====================================================================
// Fused attention, one (tb-block, head, 64-q-row tile) per workgroup.
// qkv: [MROWS, 2304] fp32 (q|k|v), amask: [NTBB, 512, 512] fp32.
// ctx: [MROWS, 768] fp32 (heads merged).
// 128 threads (4 waves, 16 q rows each). Scores kept in LDS (f32).
// =====================================================================
__global__ __launch_bounds__(128)
void k_attn(const float* __restrict__ qkv,
            const float* __restrict__ amask,
            float* __restrict__ ctx)
{
  extern __shared__ __attribute__((aligned(16))) char smem[];
  _Float16* Qs   = (_Float16*)smem;            // 64x64 f16
  _Float16* KVs  = Qs + 64 * 64;               // 64x64 f16 (K rows / V transposed)
  float*    Ss   = (float*)(KVs + 64 * 64);    // 64x512 f32 scores / exp
  float*    rinv = Ss + 64 * 512;              // 64 row inverse-sums

  const int tid = threadIdx.x, lane = tid & 31, wid = tid >> 5;
  const int bid = blockIdx.x;
  const int qt  = bid & 7;
  const int h   = (bid >> 3) % NHEAD;
  const int bt  = bid / (8 * NHEAD);
  const size_t rowbase = (size_t)bt * TLEN;
  const int q0 = qt * 64;
  const int koff_a = (lane >= 16) ? 8 : 0;
  const int koff_b = (lane >= 16) ? 16 : 0;
  const int nrow   = lane & 15;
  const int roff   = (lane >= 16) ? 8 : 0;

  // ---- load Q tile (fp32 -> f16) ----
  for (int i = tid; i < 64 * 16; i += 128) {
    int r = i >> 4, c = (i & 15) * 4;
    float4 f = *(const float4*)(qkv + (rowbase + q0 + r) * H3 + h * 64 + c);
    _Float16* d = Qs + r * 64 + c;
    d[0] = (_Float16)f.x; d[1] = (_Float16)f.y;
    d[2] = (_Float16)f.z; d[3] = (_Float16)f.w;
  }

  // ---- scores: S = Q K^T * 0.125 + mask ----
  for (int kc = 0; kc < 8; ++kc) {
    __syncthreads();
    for (int i = tid; i < 64 * 16; i += 128) {  // K chunk rows, [k'][d]
      int r = i >> 4, c = (i & 15) * 4;
      float4 f = *(const float4*)(qkv + (rowbase + kc * 64 + r) * H3 + HDIM + h * 64 + c);
      _Float16* d = KVs + r * 64 + c;
      d[0] = (_Float16)f.x; d[1] = (_Float16)f.y;
      d[2] = (_Float16)f.z; d[3] = (_Float16)f.w;
    }
    __syncthreads();
    v8f acc[4] = {};
#pragma unroll
    for (int ds = 0; ds < 2; ++ds) {
      const _Float16* pa = Qs + (wid * 16 + nrow) * 64 + ds * 32;
      v16h a = mk16(*(const v8h*)(pa + koff_a), *(const v8h*)(pa + 16 + koff_a));
#pragma unroll
      for (int tn = 0; tn < 4; ++tn) {
        const _Float16* pb = KVs + (tn * 16 + nrow) * 64 + ds * 32;
        v16h b = mk16(*(const v8h*)(pb + koff_a), *(const v8h*)(pb + 16 + koff_a));
        acc[tn] = __builtin_amdgcn_wmma_f32_16x16x32_f16(
            false, a, false, b, (short)0, acc[tn], false, false);
      }
    }
#pragma unroll
    for (int tn = 0; tn < 4; ++tn) {
      int kcol = kc * 64 + tn * 16 + nrow;
#pragma unroll
      for (int r = 0; r < 8; ++r) {
        int qr = wid * 16 + roff + r;
        float mv = (1.0f - amask[(rowbase + q0 + qr) * TLEN + kcol]) * -10000.0f;
        Ss[qr * TLEN + kcol] = acc[tn][r] * 0.125f + mv;
      }
    }
  }
  __syncthreads();

  // ---- softmax (unnormalized exp + 1/rowsum) ----
  if (tid < 64) {
    float* row = Ss + tid * TLEN;
    float mx = -3.0e38f;
    for (int j = 0; j < TLEN; ++j) mx = fmaxf(mx, row[j]);
    float s = 0.0f;
    for (int j = 0; j < TLEN; ++j) { float e = __expf(row[j] - mx); row[j] = e; s += e; }
    rinv[tid] = 1.0f / s;
  }
  __syncthreads();

  // ---- ctx = P V (scale by rinv in epilogue) ----
  v8f cacc[4] = {};
  for (int kc = 0; kc < 8; ++kc) {
    __syncthreads();
    for (int i = tid; i < 64 * 16; i += 128) {  // V chunk transposed: [d][k']
      int r = i >> 4, c = (i & 15) * 4;
      float4 f = *(const float4*)(qkv + (rowbase + kc * 64 + r) * H3 + 2 * HDIM + h * 64 + c);
      KVs[(c + 0) * 64 + r] = (_Float16)f.x;
      KVs[(c + 1) * 64 + r] = (_Float16)f.y;
      KVs[(c + 2) * 64 + r] = (_Float16)f.z;
      KVs[(c + 3) * 64 + r] = (_Float16)f.w;
    }
    __syncthreads();
#pragma unroll
    for (int ks = 0; ks < 2; ++ks) {
      const float* pa = Ss + (wid * 16 + nrow) * TLEN + kc * 64 + ks * 32;
      v16h a;
#pragma unroll
      for (int e = 0; e < 8; ++e) {
        a[e]     = (_Float16)pa[koff_a + e];
        a[e + 8] = (_Float16)pa[16 + koff_a + e];
      }
#pragma unroll
      for (int tn = 0; tn < 4; ++tn) {
        const _Float16* pb = KVs + (tn * 16 + nrow) * 64 + ks * 32;
        v16h b = mk16(*(const v8h*)(pb + koff_a), *(const v8h*)(pb + 16 + koff_a));
        cacc[tn] = __builtin_amdgcn_wmma_f32_16x16x32_f16(
            false, a, false, b, (short)0, cacc[tn], false, false);
      }
    }
  }
#pragma unroll
  for (int tn = 0; tn < 4; ++tn) {
    int d = tn * 16 + nrow;
#pragma unroll
    for (int r = 0; r < 8; ++r) {
      int qr = wid * 16 + roff + r;
      ctx[(rowbase + q0 + qr) * HDIM + h * 64 + d] = cacc[tn][r] * rinv[qr];
    }
  }
}

// =====================================================================
// elementwise / reduction helpers
// =====================================================================
__global__ void k_zero(float* p, size_t n) {
  size_t i = (size_t)blockIdx.x * blockDim.x + threadIdx.x;
  size_t st = (size_t)gridDim.x * blockDim.x;
  for (; i < n; i += st) p[i] = 0.0f;
}

// transpose-convert: src fp32 [K,N] -> dst f16 [N,K]; one block per n.
__global__ void k_cvt_t(const float* __restrict__ s, _Float16* __restrict__ d,
                        int K, int N) {
  int n = blockIdx.x;
  for (int k = threadIdx.x; k < K; k += blockDim.x)
    d[(size_t)n * K + k] = (_Float16)s[(size_t)k * N + n];
}

// packed transposed QKV: dst f16 [2304][768]; n<768 -> q, <1536 -> k, else v.
__global__ void k_pack_qkv_t(const float* __restrict__ q, const float* __restrict__ k,
                             const float* __restrict__ v, _Float16* __restrict__ d) {
  int n = blockIdx.x;                  // 0..2303 output col
  const float* src = (n < HDIM) ? q : (n < 2 * HDIM) ? k : v;
  int col = n % HDIM;
  for (int kk = threadIdx.x; kk < HDIM; kk += blockDim.x)
    d[(size_t)n * HDIM + kk] = (_Float16)src[(size_t)kk * HDIM + col];
}

__global__ void k_cat3f(const float* a, const float* b, const float* c, float* d, int n) {
  int i = blockIdx.x * blockDim.x + threadIdx.x;
  if (i < n) { d[i] = a[i]; d[n + i] = b[i]; d[2 * n + i] = c[i]; }
}

__global__ void k_gather_pos(const float* __restrict__ seq, const int* __restrict__ pos,
                             float* __restrict__ dst, int dstride) {
  int m = blockIdx.x;                 // 0..MROWS-1
  int b = m / (5 * TLEN);
  int p = pos[m];
  const float* s = seq + ((size_t)b * SEQL + p) * HDIM;
  float* d = dst + (size_t)m * dstride;
  for (int j = threadIdx.x; j < HDIM; j += blockDim.x) d[j] = s[j];
}

__global__ void k_seg_scatter(const float* __restrict__ x, int xstride,
                              const int* __restrict__ ids,
                              float* __restrict__ seg, float* __restrict__ cnt) {
  int m = blockIdx.x;
  int bt = m >> 9;
  int id = ids[m];
  float* d = seg + ((size_t)bt * MSEG + id) * HDIM;
  const float* s = x + (size_t)m * xstride;
  for (int j = threadIdx.x; j < HDIM; j += blockDim.x) atomicAdd(d + j, s[j]);
  if (cnt && threadIdx.x == 0) atomicAdd(cnt + bt * MSEG + id, 1.0f);
}

__global__ void k_seg_gather(const float* __restrict__ seg, const float* __restrict__ cnt,
                             const int* __restrict__ ids,
                             float* __restrict__ dst, int dstride) {
  int m = blockIdx.x;
  int bt = m >> 9;
  int id = ids[m];
  const float* s = seg + ((size_t)bt * MSEG + id) * HDIM;
  float sc = cnt ? (1.0f / (cnt[bt * MSEG + id] + 0.001f)) : 1.0f;
  float* d = dst + (size_t)m * dstride;
  for (int j = threadIdx.x; j < HDIM; j += blockDim.x) d[j] = s[j] * sc;
}

__global__ void k_copy_rows(const float* __restrict__ s, int ss,
                            float* __restrict__ d, int ds, int w) {
  size_t r = blockIdx.x;
  for (int j = threadIdx.x; j < w; j += blockDim.x) d[r * ds + j] = s[r * ss + j];
}

__global__ __launch_bounds__(256)
void k_add_ln(const float* __restrict__ a, const float* __restrict__ b,
              const float* __restrict__ g, const float* __restrict__ be,
              float* __restrict__ out) {
  __shared__ float sh[HDIM];
  __shared__ float red[256];
  int tid = threadIdx.x;
  size_t row = blockIdx.x;
  const float* pa = a + row * HDIM;
  const float* pb = b + row * HDIM;
  float loc = 0.0f;
  for (int j = tid; j < HDIM; j += 256) { float v = pa[j] + pb[j]; sh[j] = v; loc += v; }
  red[tid] = loc; __syncthreads();
  for (int s = 128; s > 0; s >>= 1) { if (tid < s) red[tid] += red[tid + s]; __syncthreads(); }
  float mean = red[0] * (1.0f / HDIM); __syncthreads();
  loc = 0.0f;
  for (int j = tid; j < HDIM; j += 256) { float d = sh[j] - mean; loc += d * d; }
  red[tid] = loc; __syncthreads();
  for (int s = 128; s > 0; s >>= 1) { if (tid < s) red[tid] += red[tid + s]; __syncthreads(); }
  float inv = rsqrtf(red[0] * (1.0f / HDIM) + 1e-12f);
  for (int j = tid; j < HDIM; j += 256)
    out[row * HDIM + j] = (sh[j] - mean) * inv * g[j] + be[j];
}

__global__ void k_table_scatter(const float* __restrict__ x3, const int* __restrict__ pos,
                                float* __restrict__ table, float* __restrict__ cnt) {
  int m = blockIdx.x;              // 0..MROWS-1
  int b = m / (5 * TLEN);
  int p = pos[m];
  float* d = table + ((size_t)b * SEQL + p) * H3;
  const float* s = x3 + (size_t)m * H3;
  for (int j = threadIdx.x; j < H3; j += blockDim.x) atomicAdd(d + j, s[j]);
  if (threadIdx.x == 0) atomicAdd(cnt + b * SEQL + p, 1.0f);
}

__global__ void k_build_seqtab(const float* __restrict__ seq, const float* __restrict__ table,
                               const float* __restrict__ cnt, float* __restrict__ dst) {
  size_t m = blockIdx.x;           // 0..OUTR-1
  float sc = 1.0f / (cnt[m] + 0.001f);
  float* d = dst + m * (HDIM + H3);
  const float* s0 = seq + m * HDIM;
  const float* t  = table + m * H3;
  for (int j = threadIdx.x; j < HDIM; j += blockDim.x) d[j] = s0[j];
  for (int j = threadIdx.x; j < H3; j += blockDim.x) d[HDIM + j] = t[j] * sc;
}

__global__ void k_head2(const float* __restrict__ x, int K, const float* __restrict__ w,
                        const float* __restrict__ b2,
                        float* __restrict__ so, float* __restrict__ eo) {
  int m = blockIdx.x * blockDim.x + threadIdx.x;
  if (m >= OUTR) return;
  const float* p = x + (size_t)m * K;
  float s = b2[0], e = b2[1];
  for (int j = 0; j < K; ++j) { float v = p[j]; s += v * w[2 * j]; e += v * w[2 * j + 1]; }
  so[m] = s; eo[m] = e;
}

__global__ void k_mix(const float* s1, const float* e1, const float* s2, const float* e2,
                      const int* __restrict__ cols, float* st, float* en) {
  int m = blockIdx.x * blockDim.x + threadIdx.x;
  if (m >= OUTR) return;
  float tw = (cols[m] > 0) ? 1.0f : 0.0f;
  float itw = 1.0f - tw;
  st[m] = s1[m] * itw + s2[m] * tw;
  en[m] = e1[m] * itw + e2[m] * tw;
}

__global__ __launch_bounds__(256)
void k_softmax_row(const float* __restrict__ in, float* __restrict__ out, int L) {
  __shared__ float red[256];
  int tid = threadIdx.x;
  const float* p = in + (size_t)blockIdx.x * L;
  float* o = out + (size_t)blockIdx.x * L;
  float mx = -3.0e38f;
  for (int j = tid; j < L; j += 256) mx = fmaxf(mx, p[j]);
  red[tid] = mx; __syncthreads();
  for (int s = 128; s > 0; s >>= 1) { if (tid < s) red[tid] = fmaxf(red[tid], red[tid + s]); __syncthreads(); }
  mx = red[0]; __syncthreads();
  float sum = 0.0f;
  for (int j = tid; j < L; j += 256) sum += __expf(p[j] - mx);
  red[tid] = sum; __syncthreads();
  for (int s = 128; s > 0; s >>= 1) { if (tid < s) red[tid] += red[tid + s]; __syncthreads(); }
  float inv = 1.0f / red[0];
  for (int j = tid; j < L; j += 256) o[j] = __expf(p[j] - mx) * inv;
}

// =====================================================================
// orchestration
// =====================================================================
extern "C" void kernel_launch(void* const* d_in, const int* in_sizes, int n_in,
                              void* d_out, int out_size, void* d_ws, size_t ws_size,
                              hipStream_t stream) {
  (void)in_sizes; (void)n_in; (void)out_size; (void)ws_size;

  // ---- input map (top-level dict insertion order; params alphabetically) ----
  const float* seq    = (const float*)d_in[0];
  const float* amask  = (const float*)d_in[1];
  const float* ad1_b  = (const float*)d_in[2];
  const float* ad1_w  = (const float*)d_in[3];
  const float* ad2_b  = (const float*)d_in[4];
  const float* ad2_w  = (const float*)d_in[5];
  // enc dict (alphabetical): f1b f1w f2b f2w kb kw ln1b ln1g ln2b ln2g ob ow qb qw vb vw
  const float* E1[16]; for (int i = 0; i < 16; ++i) E1[i] = (const float*)d_in[6 + i];
  const float* E2[16]; for (int i = 0; i < 16; ++i) E2[i] = (const float*)d_in[22 + i];
  const float* h1_b = (const float*)d_in[38];
  const float* h1_w = (const float*)d_in[39];
  const float* h2_b = (const float*)d_in[40];
  const float* h2_w = (const float*)d_in[41];
  const float* qa_b = (const float*)d_in[42];
  const float* qa_w = (const float*)d_in[43];
  const float* qt_b = (const float*)d_in[44];
  const float* qt_w = (const float*)d_in[45];
  const int* col_ids = (const int*)d_in[47];
  const int* pos_tb  = (const int*)d_in[48];
  const int* cids    = (const int*)d_in[49];
  const int* rids    = (const int*)d_in[50];
  float* out = (float*)d_out;

  enum { F1B, F1W, F2B, F2W, KB, KW, LN1B, LN1G, LN2B, LN2G, OB, OW, QB, QW, VB, VW };

  // ---- workspace carving ----
  char* base = (char*)d_ws;
  size_t cur = 0;
  auto alloc = [&](size_t bytes) -> void* {
    cur = (cur + 255) & ~(size_t)255;
    void* p = base + cur; cur += bytes; return p;
  };
  _Float16* WQKV1 = (_Float16*)alloc((size_t)HDIM * H3 * 2);     // [2304][768]
  _Float16* WO1   = (_Float16*)alloc((size_t)HDIM * HDIM * 2);   // [768][768]
  _Float16* WF11  = (_Float16*)alloc((size_t)HDIM * FFDIM * 2);  // [3072][768]
  _Float16* WF21  = (_Float16*)alloc((size_t)FFDIM * HDIM * 2);  // [768][3072]
  _Float16* WQKV2 = (_Float16*)alloc((size_t)HDIM * H3 * 2);
  _Float16* WO2   = (_Float16*)alloc((size_t)HDIM * HDIM * 2);
  _Float16* WF12  = (_Float16*)alloc((size_t)HDIM * FFDIM * 2);
  _Float16* WF22  = (_Float16*)alloc((size_t)FFDIM * HDIM * 2);
  _Float16* WAD1  = (_Float16*)alloc((size_t)H3 * HDIM * 2);     // [768][2304]
  _Float16* WAD2  = (_Float16*)alloc((size_t)H3 * HDIM * 2);
  _Float16* WH1   = (_Float16*)alloc((size_t)FFDIM * 1536 * 2);  // [1536][3072]
  _Float16* WH2   = (_Float16*)alloc((size_t)1536 * HDIM * 2);   // [768][1536]
  float* qkvb1 = (float*)alloc(H3 * 4);
  float* qkvb2 = (float*)alloc(H3 * 4);
  float* bx    = (float*)alloc((size_t)MROWS * HDIM * 4);
  float* by    = (float*)alloc((size_t)MROWS * HDIM * 4);
  float* bctx  = (float*)alloc((size_t)MROWS * HDIM * 4);
  float* bseg  = (float*)alloc((size_t)NTBB * MSEG * HDIM * 4);
  float* bsegc = (float*)alloc((size_t)NTBB * MSEG * 4);
  float* u2    = (float*)alloc((size_t)OUTR * (HDIM + H3) * 4);   // qkv / seqtab
  float* u1    = (float*)alloc((size_t)MROWS * FFDIM * 4);        // cat / ffn
  float* btab  = (float*)alloc((size_t)OUTR * H3 * 4);
  float* btc   = (float*)alloc((size_t)OUTR * 4);
  float* bh1   = (float*)alloc((size_t)OUTR * 1536 * 4);
  float* bh2   = (float*)alloc((size_t)OUTR * HDIM * 4);
  float* s1v = (float*)alloc(OUTR * 4);
  float* e1v = (float*)alloc(OUTR * 4);
  float* s2v = (float*)alloc(OUTR * 4);
  float* e2v = (float*)alloc(OUTR * 4);
  float* bst = (float*)alloc(OUTR * 4);
  float* ben = (float*)alloc(OUTR * 4);

  const size_t ATTN_SMEM = 64 * 64 * 2 * 2 + 64 * 512 * 4 + 64 * 4;  // 147712 B

  // ---- weight conversion to transposed f16 [N][K] ----
  k_pack_qkv_t<<<H3, 256, 0, stream>>>(E1[QW], E1[KW], E1[VW], WQKV1);
  k_pack_qkv_t<<<H3, 256, 0, stream>>>(E2[QW], E2[KW], E2[VW], WQKV2);
  k_cvt_t<<<HDIM, 256, 0, stream>>>(E1[OW], WO1, HDIM, HDIM);
  k_cvt_t<<<FFDIM, 256, 0, stream>>>(E1[F1W], WF11, HDIM, FFDIM);
  k_cvt_t<<<HDIM, 256, 0, stream>>>(E1[F2W], WF21, FFDIM, HDIM);
  k_cvt_t<<<HDIM, 256, 0, stream>>>(E2[OW], WO2, HDIM, HDIM);
  k_cvt_t<<<FFDIM, 256, 0, stream>>>(E2[F1W], WF12, HDIM, FFDIM);
  k_cvt_t<<<HDIM, 256, 0, stream>>>(E2[F2W], WF22, FFDIM, HDIM);
  k_cvt_t<<<HDIM, 256, 0, stream>>>(ad1_w, WAD1, H3, HDIM);
  k_cvt_t<<<HDIM, 256, 0, stream>>>(ad2_w, WAD2, H3, HDIM);
  k_cvt_t<<<1536, 256, 0, stream>>>(h1_w, WH1, FFDIM, 1536);
  k_cvt_t<<<HDIM, 256, 0, stream>>>(h2_w, WH2, 1536, HDIM);
  k_cat3f<<<3, 256, 0, stream>>>(E1[QB], E1[KB], E1[VB], qkvb1, HDIM);
  k_cat3f<<<3, 256, 0, stream>>>(E2[QB], E2[KB], E2[VB], qkvb2, HDIM);

  auto seg_pool = [&](const float* src, int sstride, const int* ids, float* dst, bool mean) {
    k_zero<<<2048, 256, 0, stream>>>(bseg, (size_t)NTBB * MSEG * HDIM);
    if (mean) k_zero<<<64, 256, 0, stream>>>(bsegc, (size_t)NTBB * MSEG);
    k_seg_scatter<<<MROWS, 256, 0, stream>>>(src, sstride, ids, bseg, mean ? bsegc : nullptr);
    k_seg_gather<<<MROWS, 256, 0, stream>>>(bseg, mean ? bsegc : nullptr, ids, dst, H3);
  };

  auto run_enc = [&](const _Float16* Wqkv, const float* bqkv,
                     const _Float16* Wo, const float* ob,
                     const float* ln1g, const float* ln1b,
                     const _Float16* Wf1, const float* f1b,
                     const _Float16* Wf2, const float* f2b,
                     const float* ln2g, const float* ln2b) {
    k_gemm<<<dim3(80, 18), 256, 0, stream>>>(bx, HDIM, Wqkv, bqkv, u2, H3, H3, HDIM, 0);
    k_attn<<<dim3(NTBB * NHEAD * 8), 128, ATTN_SMEM, stream>>>(u2, amask, bctx);
    k_gemm<<<dim3(80, 6), 256, 0, stream>>>(bctx, HDIM, Wo, ob, by, HDIM, HDIM, HDIM, 0);
    k_add_ln<<<MROWS, 256, 0, stream>>>(bx, by, ln1g, ln1b, bx);
    k_gemm<<<dim3(80, 24), 256, 0, stream>>>(bx, HDIM, Wf1, f1b, u1, FFDIM, FFDIM, HDIM, 1);
    k_gemm<<<dim3(80, 6), 256, 0, stream>>>(u1, FFDIM, Wf2, f2b, by, HDIM, HDIM, FFDIM, 0);
    k_add_ln<<<MROWS, 256, 0, stream>>>(bx, by, ln2g, ln2b, bx);
  };

  // ---- stage 1: gather + sum-pool + ad1 ----
  k_gather_pos<<<MROWS, 256, 0, stream>>>(seq, pos_tb, u1, H3);
  seg_pool(u1, H3, cids, u1 + HDIM, false);
  seg_pool(u1, H3, rids, u1 + 2 * HDIM, false);
  k_gemm<<<dim3(80, 6), 256, 0, stream>>>(u1, H3, WAD1, ad1_b, bx, HDIM, HDIM, H3, 1);

  // ---- encoder 1 ----
  run_enc(WQKV1, qkvb1, WO1, E1[OB], E1[LN1G], E1[LN1B],
          WF11, E1[F1B], WF21, E1[F2B], E1[LN2G], E1[LN2B]);

  // ---- stage 2: mean-pool + ad2 ----
  seg_pool(bx, HDIM, cids, u1 + HDIM, true);
  seg_pool(bx, HDIM, rids, u1 + 2 * HDIM, true);
  k_copy_rows<<<MROWS, 256, 0, stream>>>(bx, HDIM, u1, H3, HDIM);
  k_gemm<<<dim3(80, 6), 256, 0, stream>>>(u1, H3, WAD2, ad2_b, bx, HDIM, HDIM, H3, 1);

  // ---- encoder 2 ----
  run_enc(WQKV2, qkvb2, WO2, E2[OB], E2[LN1G], E2[LN1B],
          WF12, E2[F1B], WF22, E2[F2B], E2[LN2G], E2[LN2B]);

  // ---- stage 3: mean-pool -> x3 -> scatter_mean table -> heads ----
  seg_pool(bx, HDIM, cids, u1 + HDIM, true);
  seg_pool(bx, HDIM, rids, u1 + 2 * HDIM, true);
  k_copy_rows<<<MROWS, 256, 0, stream>>>(bx, HDIM, u1, H3, HDIM);

  k_zero<<<2048, 256, 0, stream>>>(btab, (size_t)OUTR * H3);
  k_zero<<<64, 256, 0, stream>>>(btc, (size_t)OUTR);
  k_table_scatter<<<MROWS, 256, 0, stream>>>(u1, pos_tb, btab, btc);
  k_build_seqtab<<<OUTR, 256, 0, stream>>>(seq, btab, btc, u2);

  k_gemm<<<dim3(64, 12), 256, 0, stream>>>(u2, HDIM + H3, WH1, h1_b, bh1, 1536, 1536, FFDIM, 1);
  k_gemm<<<dim3(64, 6), 256, 0, stream>>>(bh1, 1536, WH2, h2_b, bh2, HDIM, HDIM, 1536, 1);

  k_head2<<<(OUTR + 255) / 256, 256, 0, stream>>>(seq, HDIM, qa_w, qa_b, s1v, e1v);
  k_head2<<<(OUTR + 255) / 256, 256, 0, stream>>>(bh2, HDIM, qt_w, qt_b, s2v, e2v);
  k_mix<<<(OUTR + 255) / 256, 256, 0, stream>>>(s1v, e1v, s2v, e2v, col_ids, bst, ben);
  k_softmax_row<<<BATCH, 256, 0, stream>>>(bst, out, SEQL);
  k_softmax_row<<<BATCH, 256, 0, stream>>>(ben, out + OUTR, SEQL);
}
